// Classifier_69818988363919
// MI455X (gfx1250) — matrix operand order repaired and checked by reading, not compiled, not run on hardware
//
#include <hip/hip_runtime.h>

typedef __attribute__((ext_vector_type(16))) _Float16 v16h;
typedef __attribute__((ext_vector_type(8)))  float    v8f;

#define D_IN    64
#define D_ENC   128
#define NCLS    10
#define JWIN    30
#define SEG_PW  8                   // segments per wave
#define MT_PW   15                  // M-tiles per wave (8*30/16)
#define NT      8                   // N-tiles (128/16)
#define WAVES   8
#define SEG_PB  (SEG_PW * WAVES)    // 64 segments per 256-thread block
#define BPAD    8                   // f16 row pad: stride 36 dwords -> all 64 banks hit once

__device__ __forceinline__ v16h cvt16(const float4 a, const float4 b,
                                      const float4 c, const float4 d) {
    v16h r;
    r[0]=(_Float16)a.x;  r[1]=(_Float16)a.y;  r[2]=(_Float16)a.z;  r[3]=(_Float16)a.w;
    r[4]=(_Float16)b.x;  r[5]=(_Float16)b.y;  r[6]=(_Float16)b.z;  r[7]=(_Float16)b.w;
    r[8]=(_Float16)c.x;  r[9]=(_Float16)c.y;  r[10]=(_Float16)c.z; r[11]=(_Float16)c.w;
    r[12]=(_Float16)d.x; r[13]=(_Float16)d.y; r[14]=(_Float16)d.z; r[15]=(_Float16)d.w;
    return r;
}

// Single-instruction ReLU (avoids the compiler's canonicalize + max pair).
__device__ __forceinline__ float relu1(float x) {
    float r;
    asm("v_max_num_f32_e32 %0, 0, %1" : "=v"(r) : "v"(x));
    return r;
}

// One N-tile of one M-tile.  MT and NTI are template parameters so every
// segreg index is a frontend literal -> SROA must keep segreg in registers.
template <int MT, int NTI>
__device__ __forceinline__ void nt_step(const _Float16 (*Bh)[D_IN + BPAD],
                                        int col, int khalf,
                                        v16h afr0, v16h afr1,
                                        float (&segreg)[SEG_PW][NT])
{
    constexpr int sBase = (MT * 8) / 15;               // segment of row r=0
    constexpr int cutT  = (sBase + 1) * 15 - MT * 8;   // rows inside sBase
    constexpr int cut   = (cutT > 8) ? 8 : cutT;

    // B fragments from LDS (conflict-free: row stride = 36 dwords)
    v16h b0 = *(const v16h*)&Bh[NTI * 16 + col][khalf * 16];
    v16h b1 = *(const v16h*)&Bh[NTI * 16 + col][32 + khalf * 16];

    v8f acc = {};
    acc = __builtin_amdgcn_wmma_f32_16x16x32_f16(
              false, afr0, false, b0, (short)0, acc, false, false);
    acc = __builtin_amdgcn_wmma_f32_16x16x32_f16(
              false, afr1, false, b1, (short)0, acc, false, false);

    float v0 = relu1(acc[0]), v1 = relu1(acc[1]);
    float v2 = relu1(acc[2]), v3 = relu1(acc[3]);
    float v4 = relu1(acc[4]), v5 = relu1(acc[5]);
    float v6 = relu1(acc[6]), v7 = relu1(acc[7]);

    if constexpr (cut >= 8) {
        // uniform tile: balanced tree reduction
        segreg[sBase][NTI] +=
            ((v0 + v1) + (v2 + v3)) + ((v4 + v5) + (v6 + v7));
    } else {
        // boundary tile: two compile-time partial sums (-0.0f folds 1st add)
        float a = -0.0f, b = -0.0f;
        if constexpr (cut > 0) a += v0; else b += v0;
        if constexpr (cut > 1) a += v1; else b += v1;
        if constexpr (cut > 2) a += v2; else b += v2;
        if constexpr (cut > 3) a += v3; else b += v3;
        if constexpr (cut > 4) a += v4; else b += v4;
        if constexpr (cut > 5) a += v5; else b += v5;
        if constexpr (cut > 6) a += v6; else b += v6;
        if constexpr (cut > 7) a += v7; else b += v7;
        segreg[sBase][NTI]     += a;
        segreg[sBase + 1][NTI] += b;
    }
}

// One M-tile (16 rows) of the wave's 240-row slice.
template <int MT>
__device__ __forceinline__ void mt_step(const float* __restrict__ x,
                                        size_t rowBase, int m7, int mh,
                                        int col, int khalf,
                                        const _Float16 (*Bh)[D_IN + BPAD],
                                        float (&segreg)[SEG_PW][NT])
{
    // Row permutation: pair p = MT*8 + r -> segment s = p/15,
    // rows s*30 + p%15 (+15 for the upper M-half).
    const int p = MT * 8 + m7;
    const int s = p / 15;
    const int rowInWave = s * 15 + p + mh * 15;     // == s*30 + p%15 + mh*15
    const float* xr = x + (rowBase + (size_t)rowInWave) * D_IN + khalf * 16;

    // A fragments (contiguous k = 16*khalf + s), phased to cap transient pressure
    float4 q0 = *(const float4*)(xr);        // ks=0 : K elems 0..7
    float4 q1 = *(const float4*)(xr + 4);
    float4 q2 = *(const float4*)(xr + 8);    // ks=0 : K elems 8..15
    float4 q3 = *(const float4*)(xr + 12);
    v16h afr0 = cvt16(q0, q1, q2, q3);
    float4 q4 = *(const float4*)(xr + 32);   // ks=1
    float4 q5 = *(const float4*)(xr + 36);
    float4 q6 = *(const float4*)(xr + 40);
    float4 q7 = *(const float4*)(xr + 44);
    v16h afr1 = cvt16(q4, q5, q6, q7);

    nt_step<MT, 0>(Bh, col, khalf, afr0, afr1, segreg);
    nt_step<MT, 1>(Bh, col, khalf, afr0, afr1, segreg);
    nt_step<MT, 2>(Bh, col, khalf, afr0, afr1, segreg);
    nt_step<MT, 3>(Bh, col, khalf, afr0, afr1, segreg);
    nt_step<MT, 4>(Bh, col, khalf, afr0, afr1, segreg);
    nt_step<MT, 5>(Bh, col, khalf, afr0, afr1, segreg);
    nt_step<MT, 6>(Bh, col, khalf, afr0, afr1, segreg);
    nt_step<MT, 7>(Bh, col, khalf, afr0, afr1, segreg);
}

__global__ __launch_bounds__(256, 2)
void cls_fused_kernel(const float* __restrict__ x,
                      const float* __restrict__ Wloc,
                      const float* __restrict__ W,
                      float* __restrict__ out)
{
    __shared__ _Float16 Bh[D_ENC][D_IN + BPAD];       // Wloc as f16, 18.4 KB
    __shared__ float    ldsseg[WAVES][SEG_PW][D_ENC]; // 32 KB, per-wave regions

    const int tid   = threadIdx.x;
    const int wave  = tid >> 5;
    const int lane  = tid & 31;
    const int col   = lane & 15;     // A: M index ; B/C/D: N index
    const int khalf = lane >> 4;     // K-chunk half for A/B operands
    const int m7    = col & 7;       // C-layout VGPR row within half
    const int mh    = col >> 3;      // which M-half this A row feeds

    const int    waveSeg = blockIdx.x * SEG_PB + wave * SEG_PW;
    const size_t rowBase = (size_t)waveSeg * JWIN;

    // ---- stage Wloc (f32 -> f16) into LDS, amortized over all 8 waves ----
    for (int i = tid; i < D_ENC * D_IN; i += 256) {
        int e = i >> 6, d = i & 63;
        Bh[e][d] = (_Float16)Wloc[i];
    }
    // zero this wave's LDS accumulator (own region; DS ops are wave-ordered)
    {
        float4 z = make_float4(0.f, 0.f, 0.f, 0.f);
        float4* zp = (float4*)&ldsseg[wave][0][0];
#pragma unroll
        for (int i = 0; i < (SEG_PW * D_ENC / 4) / 32; ++i)
            zp[lane + 32 * i] = z;
    }
    __syncthreads();

    // ---- register segment accumulators; -0.0f so the first fadd folds ----
    float segreg[SEG_PW][NT];
#pragma unroll
    for (int s = 0; s < SEG_PW; ++s)
#pragma unroll
        for (int nt = 0; nt < NT; ++nt) segreg[s][nt] = -0.0f;

    // ---- main loop: 15 M-tiles, explicit template instantiations ----
    mt_step< 0>(x, rowBase, m7, mh, col, khalf, Bh, segreg);
    mt_step< 1>(x, rowBase, m7, mh, col, khalf, Bh, segreg);
    mt_step< 2>(x, rowBase, m7, mh, col, khalf, Bh, segreg);
    mt_step< 3>(x, rowBase, m7, mh, col, khalf, Bh, segreg);
    mt_step< 4>(x, rowBase, m7, mh, col, khalf, Bh, segreg);
    mt_step< 5>(x, rowBase, m7, mh, col, khalf, Bh, segreg);
    mt_step< 6>(x, rowBase, m7, mh, col, khalf, Bh, segreg);
    mt_step< 7>(x, rowBase, m7, mh, col, khalf, Bh, segreg);
    mt_step< 8>(x, rowBase, m7, mh, col, khalf, Bh, segreg);
    mt_step< 9>(x, rowBase, m7, mh, col, khalf, Bh, segreg);
    mt_step<10>(x, rowBase, m7, mh, col, khalf, Bh, segreg);
    mt_step<11>(x, rowBase, m7, mh, col, khalf, Bh, segreg);
    mt_step<12>(x, rowBase, m7, mh, col, khalf, Bh, segreg);
    mt_step<13>(x, rowBase, m7, mh, col, khalf, Bh, segreg);
    mt_step<14>(x, rowBase, m7, mh, col, khalf, Bh, segreg);

    // ---- merge the two lane-halves' partials into LDS (once per wave) ----
#pragma unroll
    for (int s = 0; s < SEG_PW; ++s)
#pragma unroll
        for (int nt = 0; nt < NT; ++nt)
            atomicAdd(&ldsseg[wave][s][nt * 16 + col], segreg[s][nt]);

    // ---- classifier: out[s,c] = (segsum[s] . W[c]) / J  (80 outputs/wave) --
    for (int pp = lane; pp < SEG_PW * NCLS; pp += 32) {
        int s = pp / NCLS, c = pp % NCLS;
        const float* wr = W + c * D_ENC;
        float a = 0.0f;
#pragma unroll 8
        for (int e = 0; e < D_ENC; e += 4) {
            float4 f = *(const float4*)&ldsseg[wave][s][e];
            float4 g = *(const float4*)(wr + e);
            a += f.x * g.x + f.y * g.y + f.z * g.z + f.w * g.w;
        }
        out[(size_t)(waveSeg + s) * NCLS + c] = a * (1.0f / JWIN);
    }
}

extern "C" void kernel_launch(void* const* d_in, const int* in_sizes, int n_in,
                              void* d_out, int out_size, void* d_ws, size_t ws_size,
                              hipStream_t stream) {
    const float* x    = (const float*)d_in[0];   // [1200000, 64]
    const float* Wloc = (const float*)d_in[1];   // [128, 64]
    const float* W    = (const float*)d_in[2];   // [10, 128]
    float* out        = (float*)d_out;           // [40000, 10]

    const int L        = in_sizes[0] / D_IN;     // 1,200,000
    const int segments = L / JWIN;               // 40,000
    const int blocks   = segments / SEG_PB;      // 625

    cls_fused_kernel<<<blocks, 256, 0, stream>>>(x, Wloc, W, out);
}
